// TransformerEncoderLayer_87402584473644
// MI455X (gfx1250) — matrix operand order repaired and checked by reading, not compiled, not run on hardware
//
#include <hip/hip_runtime.h>

// ---------------------------------------------------------------------------
// CDNA5 WMMA types / helpers
// ---------------------------------------------------------------------------
typedef __attribute__((ext_vector_type(16))) __bf16 v16bf;
typedef __attribute__((ext_vector_type(8)))  float  v8f;

union Frag { uint4 u[2]; v16bf v; };

static __device__ __forceinline__ v8f wmma_bf16(const Frag& a, const Frag& b, v8f c) {
    return __builtin_amdgcn_wmma_f32_16x16x32_bf16(
        false, a.v, false, b.v, (short)0, c, false, false);
}

static __device__ __forceinline__ unsigned short f2bf(float f) {
    union { float f; unsigned u; } x; x.f = f;
    unsigned r = x.u + 0x7FFFu + ((x.u >> 16) & 1u);   // round-to-nearest-even
    return (unsigned short)(r >> 16);
}
static __device__ __forceinline__ unsigned pack2bf(float a, float b) {
    return (unsigned)f2bf(a) | ((unsigned)f2bf(b) << 16);
}

// ---- DPP16 row_ror rotate-allreduce within each 16-lane half (no LDS) ------
template<int CTRL>
static __device__ __forceinline__ float dpp_mov(float v) {
    return __builtin_bit_cast(float, __builtin_amdgcn_update_dpp(
        0, __builtin_bit_cast(int, v), CTRL, 0xF, 0xF, true));
}
static __device__ __forceinline__ float rowmax16(float v) {
    v = fmaxf(v, dpp_mov<0x121>(v));   // row_ror:1
    v = fmaxf(v, dpp_mov<0x122>(v));   // row_ror:2
    v = fmaxf(v, dpp_mov<0x124>(v));   // row_ror:4
    v = fmaxf(v, dpp_mov<0x128>(v));   // row_ror:8
    return v;
}
static __device__ __forceinline__ float rowsum16(float v) {
    v += dpp_mov<0x121>(v);
    v += dpp_mov<0x122>(v);
    v += dpp_mov<0x124>(v);
    v += dpp_mov<0x128>(v);
    return v;
}
static __device__ __forceinline__ float allsum32(float v) {
    v = rowsum16(v);
    v += __shfl_xor(v, 16, 32);        // single cross-half exchange
    return v;
}

// ---- async global->LDS copy path (gfx1250), guarded ------------------------
#if defined(__has_builtin)
#if __has_builtin(__builtin_amdgcn_global_load_async_to_lds_b128) && \
    __has_builtin(__builtin_amdgcn_s_wait_asynccnt)
#define USE_ASYNC_LDS 1
#endif
#endif
#ifndef USE_ASYNC_LDS
#define USE_ASYNC_LDS 0
#endif

#if USE_ASYNC_LDS
// exact param types per hipcc diagnostics:
//   arg0: 'int __vector(4) *' (generic), arg1: 'int __vector(4) __shared__ *'
typedef int vi4 __attribute__((vector_size(16)));
typedef __attribute__((address_space(3))) vi4* lvi4_t;
#endif

#define S_LEN 1024
#define DMODEL 512
#define NHEAD 8
#define HDIM 64
#define TTOK 16384   // B*S

// ---------------------------------------------------------------------------
// Weight convert+transpose: W[l][k][n] (f32) -> Wt[l][n][k] (bf16), once.
// ---------------------------------------------------------------------------
__global__ __launch_bounds__(256)
void cvtT_kernel(const float* __restrict__ W, unsigned short* __restrict__ Wt,
                 int lgN, int lgK)
{
    const long idx  = (long)blockIdx.x * 256 + threadIdx.x;
    const int lgKN  = lgN + lgK;
    const long l    = idx >> lgKN;
    const long r    = idx & ((1L << lgKN) - 1);
    const long k    = r >> lgN;
    const long n    = r & ((1L << lgN) - 1);
    Wt[(l << lgKN) + (n << lgK) + k] = f2bf(W[idx]);
}

// ---------------------------------------------------------------------------
// LayerNorm: one wave32 per 512-wide row; writes f32 and bf16 copies
// ---------------------------------------------------------------------------
__global__ __launch_bounds__(256)
void ln_kernel(const float* __restrict__ x, const float* __restrict__ g,
               const float* __restrict__ b, float* __restrict__ out,
               unsigned short* __restrict__ outb)
{
    const int lane = threadIdx.x & 31;
    const int w    = threadIdx.x >> 5;
    const long row = (long)blockIdx.x * 8 + w;
    const float* xr = x + row * DMODEL;

    float4 v[4];
    float s = 0.f;
    #pragma unroll
    for (int i = 0; i < 4; ++i) {
        v[i] = *(const float4*)(xr + lane * 16 + i * 4);
        s += v[i].x + v[i].y + v[i].z + v[i].w;
    }
    s = allsum32(s);
    const float mean = s * (1.f / DMODEL);

    float vs = 0.f;
    #pragma unroll
    for (int i = 0; i < 4; ++i) {
        float dx;
        dx = v[i].x - mean; vs += dx * dx;
        dx = v[i].y - mean; vs += dx * dx;
        dx = v[i].z - mean; vs += dx * dx;
        dx = v[i].w - mean; vs += dx * dx;
    }
    vs = allsum32(vs);
    const float inv = rsqrtf(vs * (1.f / DMODEL) + 1e-6f);

    float* orow = out + row * DMODEL;
    unsigned short* orb = outb + row * DMODEL;
    #pragma unroll
    for (int i = 0; i < 4; ++i) {
        const int col = lane * 16 + i * 4;
        const float4 gg = *(const float4*)(g + col);
        const float4 bb = *(const float4*)(b + col);
        float4 o;
        o.x = (v[i].x - mean) * inv * gg.x + bb.x;
        o.y = (v[i].y - mean) * inv * gg.y + bb.y;
        o.z = (v[i].z - mean) * inv * gg.z + bb.z;
        o.w = (v[i].w - mean) * inv * gg.w + bb.w;
        *(float4*)(orow + col) = o;
        uint2 ob; ob.x = pack2bf(o.x, o.y); ob.y = pack2bf(o.z, o.w);
        *(uint2*)(orb + col) = ob;
    }
}

// ---------------------------------------------------------------------------
// WMMA GEMM, all-bf16 operands, double-buffered LDS pipeline.
// Staging uses GLOBAL_LOAD_ASYNC_TO_LDS_B128 (ASYNCcnt) when available,
// else a register double-buffer.
// ---------------------------------------------------------------------------
template<bool RELU, bool HAS_BIAS, bool HAS_RES, bool OUTF, bool OUTB>
__global__ __launch_bounds__(256)
void gemm_t(const unsigned short* __restrict__ A,
            const unsigned short* __restrict__ Wt,
            const int* __restrict__ lang, long wstride,
            const float* __restrict__ bias, int bias_lstride,
            const float* __restrict__ resid,
            float* __restrict__ outf, unsigned short* __restrict__ outb,
            float scale, int N, int K)
{
    __shared__ unsigned short As[2][128][40];   // [buf][m][k] (+8 pad)
    __shared__ unsigned short Bs[2][64][40];    // [buf][n][k] (+8 pad)

    const int tid  = threadIdx.x;
    const int row0 = blockIdx.y * 128;
    const int col0 = blockIdx.x * 64;

    const int li = lang ? lang[row0 >> 10] : 0;
    const unsigned short* W = Wt + (long)li * wstride;

    const int lane = tid & 31;
    const int w    = tid >> 5;      // 0..7
    const int wr   = w & 3;         // 32-row band
    const int wc   = w >> 2;        // 32-col band
    const int half = lane >> 4;
    const int l16  = lane & 15;

    // staging indices (fixed per thread)
    const int ar0 = tid >> 2;
    const int akc = (tid & 3) << 3;
    const int bn  = tid >> 2;
    const int bkc = (tid & 3) << 3;

    const int TSTEPS = K >> 5;
    v8f acc[2][2] = {};

#if USE_ASYNC_LDS
    auto issueAsync = [&](int buf, int k0) {
        __builtin_amdgcn_global_load_async_to_lds_b128(
            (vi4*)(A + (long)(row0 + ar0) * K + k0 + akc),
            (lvi4_t)&As[buf][ar0][akc], 0, 0);
        __builtin_amdgcn_global_load_async_to_lds_b128(
            (vi4*)(A + (long)(row0 + ar0 + 64) * K + k0 + akc),
            (lvi4_t)&As[buf][ar0 + 64][akc], 0, 0);
        __builtin_amdgcn_global_load_async_to_lds_b128(
            (vi4*)(W + (long)(col0 + bn) * K + k0 + bkc),
            (lvi4_t)&Bs[buf][bn][bkc], 0, 0);
    };
    issueAsync(0, 0);
    __builtin_amdgcn_s_wait_asynccnt(0);
    __syncthreads();
#else
    uint4 ra0, ra1, rb;
    auto loadG = [&](int k0) {
        ra0 = *(const uint4*)(A + (long)(row0 + ar0) * K + k0 + akc);
        ra1 = *(const uint4*)(A + (long)(row0 + ar0 + 64) * K + k0 + akc);
        rb  = *(const uint4*)(W + (long)(col0 + bn) * K + k0 + bkc);
    };
    auto storeL = [&](int buf) {
        *(uint4*)&As[buf][ar0][akc]      = ra0;
        *(uint4*)&As[buf][ar0 + 64][akc] = ra1;
        *(uint4*)&Bs[buf][bn][bkc]       = rb;
    };
    loadG(0);
    storeL(0);
    __syncthreads();
#endif

    for (int it = 0; it < TSTEPS; ++it) {
        const int cur = it & 1;
#if USE_ASYNC_LDS
        // async copy of next tile overlaps with this tile's WMMAs;
        // buf cur^1 was last read at iter it-1, separated by that iter's barrier
        if (it + 1 < TSTEPS) issueAsync(cur ^ 1, (it + 1) << 5);
#else
        if (it + 1 < TSTEPS) loadG((it + 1) << 5);
#endif

        Frag a0, a1, b0, b1;
        const int m0 = wr * 32 + l16;
        a0.u[0] = *(const uint4*)&As[cur][m0][half * 8];
        a0.u[1] = *(const uint4*)&As[cur][m0][16 + half * 8];
        a1.u[0] = *(const uint4*)&As[cur][m0 + 16][half * 8];
        a1.u[1] = *(const uint4*)&As[cur][m0 + 16][16 + half * 8];
        const int n0 = wc * 32 + l16;
        b0.u[0] = *(const uint4*)&Bs[cur][n0][half * 8];
        b0.u[1] = *(const uint4*)&Bs[cur][n0][16 + half * 8];
        b1.u[0] = *(const uint4*)&Bs[cur][n0 + 16][half * 8];
        b1.u[1] = *(const uint4*)&Bs[cur][n0 + 16][16 + half * 8];

        acc[0][0] = wmma_bf16(a0, b0, acc[0][0]);
        acc[0][1] = wmma_bf16(a0, b1, acc[0][1]);
        acc[1][0] = wmma_bf16(a1, b0, acc[1][0]);
        acc[1][1] = wmma_bf16(a1, b1, acc[1][1]);

        if (it + 1 < TSTEPS) {
#if USE_ASYNC_LDS
            __builtin_amdgcn_s_wait_asynccnt(0);
            __syncthreads();
#else
            __syncthreads();
            storeL(cur ^ 1);
            __syncthreads();
#endif
        }
    }

    const float* bi = HAS_BIAS ? (bias + (long)li * bias_lstride) : (const float*)0;
    #pragma unroll
    for (int mi = 0; mi < 2; ++mi) {
        #pragma unroll
        for (int nj = 0; nj < 2; ++nj) {
            const int col = col0 + wc * 32 + nj * 16 + l16;
            const float bsum = HAS_BIAS ? bi[col] : 0.f;
            #pragma unroll
            for (int r = 0; r < 8; ++r) {
                const int row = row0 + wr * 32 + mi * 16 + r + 8 * half;
                float vv = acc[mi][nj][r] + bsum;
                if (RELU) vv = fmaxf(vv, 0.f);
                if (HAS_RES) vv += resid[(long)row * N + col];
                vv *= scale;
                const long o = (long)row * N + col;
                if (OUTF) outf[o] = vv;
                if (OUTB) outb[o] = f2bf(vv);
            }
        }
    }
}

// ---------------------------------------------------------------------------
// Flash attention: grid (S/64, B*H), 128 threads = 4 waves, wave = 16 q-rows.
// ---------------------------------------------------------------------------
__global__ __launch_bounds__(128)
void attn_kernel(const unsigned short* __restrict__ qb,
                 const unsigned short* __restrict__ kb,
                 const unsigned short* __restrict__ vb,
                 const unsigned char* __restrict__ mask,
                 unsigned short* __restrict__ ctx)
{
    __shared__ unsigned short Vt[64][40];      // [d][k] transposed V tile
    __shared__ unsigned short Pw[4][16][40];   // per-wave P (16 rows x 32 keys)

    const int tid  = threadIdx.x;
    const int lane = tid & 31;
    const int w    = tid >> 5;
    const int half = lane >> 4;
    const int l16  = lane & 15;

    const int bh = blockIdx.y;
    const int b  = bh >> 3;
    const int h  = bh & 7;
    const int q0 = blockIdx.x * 64 + w * 16;
    const long base = (long)b * S_LEN * DMODEL + h * HDIM;

    Frag qf0, qf1;
    {
        const unsigned short* qp = qb + base + (long)(q0 + l16) * DMODEL;
        qf0.u[0] = *(const uint4*)(qp + half * 8);
        qf0.u[1] = *(const uint4*)(qp + 16 + half * 8);
        qf1.u[0] = *(const uint4*)(qp + 32 + half * 8);
        qf1.u[1] = *(const uint4*)(qp + 32 + 16 + half * 8);
    }

    float mi[8], lsum[8];
    v8f O[4] = {};
    #pragma unroll
    for (int r = 0; r < 8; ++r) { mi[r] = -3.0e38f; lsum[r] = 0.f; }

    for (int kt = 0; kt < S_LEN; kt += 32) {
        __syncthreads();
        #pragma unroll
        for (int i = 0; i < 4; ++i) {
            const int c   = tid + i * 128;
            const int key = c >> 4;
            const int d4  = (c & 15) << 2;
            const uint2 vv = *(const uint2*)(vb + base + (long)(kt + key) * DMODEL + d4);
            Vt[d4 + 0][key] = (unsigned short)(vv.x & 0xffffu);
            Vt[d4 + 1][key] = (unsigned short)(vv.x >> 16);
            Vt[d4 + 2][key] = (unsigned short)(vv.y & 0xffffu);
            Vt[d4 + 3][key] = (unsigned short)(vv.y >> 16);
        }
        __syncthreads();

        v8f S0 = {}, S1 = {};
        #pragma unroll
        for (int sub = 0; sub < 2; ++sub) {
            const int key0 = kt + sub * 16;
            Frag kf0, kf1;
            const unsigned short* kp = kb + base + (long)(key0 + l16) * DMODEL;
            kf0.u[0] = *(const uint4*)(kp + half * 8);
            kf0.u[1] = *(const uint4*)(kp + 16 + half * 8);
            kf1.u[0] = *(const uint4*)(kp + 32 + half * 8);
            kf1.u[1] = *(const uint4*)(kp + 32 + 16 + half * 8);
            v8f s = {};
            s = wmma_bf16(qf0, kf0, s);
            s = wmma_bf16(qf1, kf1, s);
            const unsigned char mk = mask[b * S_LEN + key0 + l16];
            if (!mk) {
                #pragma unroll
                for (int r = 0; r < 8; ++r) s[r] = -1e30f;
            }
            if (sub == 0) S0 = s; else S1 = s;
        }

        #pragma unroll
        for (int r = 0; r < 8; ++r) {
            const float m2 = rowmax16(fmaxf(S0[r], S1[r]));
            const float nm = fmaxf(mi[r], m2);
            const float al = __expf(mi[r] - nm);
            const float p0 = __expf(S0[r] - nm);
            const float p1 = __expf(S1[r] - nm);
            const float rs = rowsum16(p0 + p1);
            lsum[r] = lsum[r] * al + rs;
            mi[r]   = nm;
            O[0][r] *= al; O[1][r] *= al; O[2][r] *= al; O[3][r] *= al;
            Pw[w][r + 8 * half][l16]      = f2bf(p0);
            Pw[w][r + 8 * half][16 + l16] = f2bf(p1);
        }

        Frag pf;
        pf.u[0] = *(const uint4*)&Pw[w][l16][half * 8];
        pf.u[1] = *(const uint4*)&Pw[w][l16][16 + half * 8];
        #pragma unroll
        for (int j = 0; j < 4; ++j) {
            Frag vf;
            const int dv = j * 16 + l16;
            vf.u[0] = *(const uint4*)&Vt[dv][half * 8];
            vf.u[1] = *(const uint4*)&Vt[dv][16 + half * 8];
            O[j] = wmma_bf16(pf, vf, O[j]);
        }
    }

    #pragma unroll
    for (int j = 0; j < 4; ++j) {
        #pragma unroll
        for (int r = 0; r < 8; ++r) {
            const int row = q0 + r + 8 * half;
            ctx[base + (long)row * DMODEL + j * 16 + l16] = f2bf(O[j][r] / lsum[r]);
        }
    }
}

// ---------------------------------------------------------------------------
// gate[t] = sigmoid( dot(g1[t,:], g2w[lang[b],:,0]) + g2b[lang[b]] )
// ---------------------------------------------------------------------------
__global__ __launch_bounds__(256)
void gate_kernel(const float* __restrict__ g1, const float* __restrict__ g2w,
                 const float* __restrict__ g2b, const int* __restrict__ lang,
                 float* __restrict__ gate)
{
    const int lane = threadIdx.x & 31;
    const int w    = threadIdx.x >> 5;
    const long t   = (long)blockIdx.x * 8 + w;
    const int li   = lang[t >> 10];
    const float* a  = g1 + t * DMODEL + lane * 16;
    const float* wv = g2w + (long)li * DMODEL + lane * 16;
    float s = 0.f;
    #pragma unroll
    for (int i = 0; i < 16; i += 4) {
        const float4 av  = *(const float4*)(a + i);
        const float4 wv4 = *(const float4*)(wv + i);
        s += av.x * wv4.x + av.y * wv4.y + av.z * wv4.z + av.w * wv4.w;
    }
    s = allsum32(s);
    if (lane == 0) {
        s += g2b[li];
        gate[t] = 1.f / (1.f + __expf(-s));
    }
}

// ---------------------------------------------------------------------------
// out = share*(1-g) + langs*g + resid
// ---------------------------------------------------------------------------
__global__ __launch_bounds__(256)
void combine_kernel(const float* __restrict__ share, const float* __restrict__ langs,
                    const float* __restrict__ gate, const float* __restrict__ resid,
                    float* __restrict__ out)
{
    const long i = ((long)blockIdx.x * 256 + threadIdx.x) * 4;
    const float g = gate[i >> 9];
    const float4 sh = *(const float4*)(share + i);
    const float4 la = *(const float4*)(langs + i);
    const float4 re = *(const float4*)(resid + i);
    float4 o;
    o.x = sh.x * (1.f - g) + la.x * g + re.x;
    o.y = sh.y * (1.f - g) + la.y * g + re.y;
    o.z = sh.z * (1.f - g) + la.z * g + re.z;
    o.w = sh.w * (1.f - g) + la.w * g + re.w;
    *(float4*)(out + i) = o;
}

// ---------------------------------------------------------------------------
// Orchestration
// ---------------------------------------------------------------------------
extern "C" void kernel_launch(void* const* d_in, const int* in_sizes, int n_in,
                              void* d_out, int out_size, void* d_ws, size_t ws_size,
                              hipStream_t stream)
{
    const float* x     = (const float*)d_in[0];
    const float* ln1_g = (const float*)d_in[1];
    const float* ln1_b = (const float*)d_in[2];
    const float* ln2_g = (const float*)d_in[3];
    const float* ln2_b = (const float*)d_in[4];
    const float* wq    = (const float*)d_in[5];
    const float* bq    = (const float*)d_in[6];
    const float* wk    = (const float*)d_in[7];
    const float* bk    = (const float*)d_in[8];
    const float* wv    = (const float*)d_in[9];
    const float* bv    = (const float*)d_in[10];
    const float* wo    = (const float*)d_in[11];
    const float* bo    = (const float*)d_in[12];
    const float* ff_w1 = (const float*)d_in[13];
    const float* ff_b1 = (const float*)d_in[14];
    const float* ff_w2 = (const float*)d_in[15];
    const float* ff_b2 = (const float*)d_in[16];
    const float* a_sw  = (const float*)d_in[17];
    const float* a_lw  = (const float*)d_in[18];
    const float* a_g1w = (const float*)d_in[19];
    const float* a_g1b = (const float*)d_in[20];
    const float* a_g2w = (const float*)d_in[21];
    const float* a_g2b = (const float*)d_in[22];
    const float* f_sw  = (const float*)d_in[23];
    const float* f_lw  = (const float*)d_in[24];
    const float* f_g1w = (const float*)d_in[25];
    const float* f_g1b = (const float*)d_in[26];
    const float* f_g2w = (const float*)d_in[27];
    const float* f_g2b = (const float*)d_in[28];
    const unsigned char* mask = (const unsigned char*)d_in[29];
    const int* lang    = (const int*)d_in[30];

    const long TD  = (long)TTOK * DMODEL;
    const long TFF = (long)TTOK * 2048;
    const long DD  = (long)DMODEL * DMODEL;
    const long DFF = (long)DMODEL * 2048;

    char* p = (char*)d_ws;
    float*          xn    = (float*)p;          p += TD * 4;
    unsigned short* xnb   = (unsigned short*)p; p += TD * 2;
    unsigned short* qbuf  = (unsigned short*)p; p += TD * 2;   // \ x2 (f32) overlays
    unsigned short* kbuf  = (unsigned short*)p; p += TD * 2;   // /  qbuf+kbuf
    unsigned short* vbuf  = (unsigned short*)p; p += TD * 2;   // \ x2n (f32) overlays
    unsigned short* ctxb  = (unsigned short*)p; p += TD * 2;   // /  vbuf+ctxb
    unsigned short* hb    = (unsigned short*)p; p += TD * 2;   // reused as ffh(bf16)
    unsigned short* x2nb  = (unsigned short*)p; p += TD * 2;
    float*          shrb  = (float*)p;          p += TD * 4;
    float*          lngb  = (float*)p;          p += TD * 4;
    float*          g1buf = (float*)p;          p += TD * 4;
    float*          gateb = (float*)p;          p += (long)TTOK * 4;
    unsigned short* ffh1b = (unsigned short*)p; p += TFF * 2;
    unsigned short* wtq   = (unsigned short*)p; p += DD * 2;
    unsigned short* wtk   = (unsigned short*)p; p += DD * 2;
    unsigned short* wtv   = (unsigned short*)p; p += DD * 2;
    unsigned short* wto   = (unsigned short*)p; p += DD * 2;
    unsigned short* wtasw = (unsigned short*)p; p += DD * 2;
    unsigned short* wtfsw = (unsigned short*)p; p += DD * 2;
    unsigned short* wtalw = (unsigned short*)p; p += DD * 10 * 2;
    unsigned short* wtag1 = (unsigned short*)p; p += DD * 10 * 2;
    unsigned short* wtflw = (unsigned short*)p; p += DD * 10 * 2;
    unsigned short* wtfg1 = (unsigned short*)p; p += DD * 10 * 2;
    unsigned short* wtff1 = (unsigned short*)p; p += DFF * 2;
    unsigned short* wtff2 = (unsigned short*)p; p += DFF * 2;

    float* x2   = (float*)qbuf;
    float* x2n  = (float*)vbuf;
    unsigned short* ffhb = hb;
    float* outp = (float*)d_out;

    cvtT_kernel<<<DD / 256, 256, 0, stream>>>(wq, wtq, 9, 9);
    cvtT_kernel<<<DD / 256, 256, 0, stream>>>(wk, wtk, 9, 9);
    cvtT_kernel<<<DD / 256, 256, 0, stream>>>(wv, wtv, 9, 9);
    cvtT_kernel<<<DD / 256, 256, 0, stream>>>(wo, wto, 9, 9);
    cvtT_kernel<<<DD / 256, 256, 0, stream>>>(a_sw, wtasw, 9, 9);
    cvtT_kernel<<<DD / 256, 256, 0, stream>>>(f_sw, wtfsw, 9, 9);
    cvtT_kernel<<<DD * 10 / 256, 256, 0, stream>>>(a_lw, wtalw, 9, 9);
    cvtT_kernel<<<DD * 10 / 256, 256, 0, stream>>>(a_g1w, wtag1, 9, 9);
    cvtT_kernel<<<DD * 10 / 256, 256, 0, stream>>>(f_lw, wtflw, 9, 9);
    cvtT_kernel<<<DD * 10 / 256, 256, 0, stream>>>(f_g1w, wtfg1, 9, 9);
    cvtT_kernel<<<DFF / 256, 256, 0, stream>>>(ff_w1, wtff1, 11, 9);
    cvtT_kernel<<<DFF / 256, 256, 0, stream>>>(ff_w2, wtff2, 9, 11);

    const dim3 g512(DMODEL / 64, TTOK / 128);
    const dim3 gFF(2048 / 64, TTOK / 128);

    ln_kernel<<<TTOK / 8, 256, 0, stream>>>(x, ln1_g, ln1_b, xn, xnb);
    gemm_t<false,true,false,false,true><<<g512, 256, 0, stream>>>(
        xnb, wtq, nullptr, 0, bq, 0, nullptr, nullptr, qbuf, 0.125f, 512, 512);
    gemm_t<false,true,false,false,true><<<g512, 256, 0, stream>>>(
        xnb, wtk, nullptr, 0, bk, 0, nullptr, nullptr, kbuf, 1.f, 512, 512);
    gemm_t<false,true,false,false,true><<<g512, 256, 0, stream>>>(
        xnb, wtv, nullptr, 0, bv, 0, nullptr, nullptr, vbuf, 1.f, 512, 512);
    attn_kernel<<<dim3(S_LEN / 64, 16 * NHEAD), 128, 0, stream>>>(qbuf, kbuf, vbuf, mask, ctxb);
    gemm_t<false,true,false,false,true><<<g512, 256, 0, stream>>>(
        ctxb, wto, nullptr, 0, bo, 0, nullptr, nullptr, hb, 1.f, 512, 512);
    gemm_t<false,false,false,true,false><<<g512, 256, 0, stream>>>(
        hb, wtasw, nullptr, 0, nullptr, 0, nullptr, shrb, nullptr, 1.f, 512, 512);
    gemm_t<false,false,false,true,false><<<g512, 256, 0, stream>>>(
        hb, wtalw, lang, DD, nullptr, 0, nullptr, lngb, nullptr, 1.f, 512, 512);
    gemm_t<true,true,true,true,false><<<g512, 256, 0, stream>>>(
        xnb, wtag1, lang, DD, a_g1b, 512, xn, g1buf, nullptr, 1.f, 512, 512);
    gate_kernel<<<TTOK / 8, 256, 0, stream>>>(g1buf, a_g2w, a_g2b, lang, gateb);
    combine_kernel<<<TD / 1024, 256, 0, stream>>>(shrb, lngb, gateb, xn, x2);
    ln_kernel<<<TTOK / 8, 256, 0, stream>>>(x2, ln2_g, ln2_b, x2n, x2nb);
    gemm_t<true,true,false,false,true><<<gFF, 256, 0, stream>>>(
        x2nb, wtff1, nullptr, 0, ff_b1, 0, nullptr, nullptr, ffh1b, 1.f, 2048, 512);
    gemm_t<false,true,false,false,true><<<g512, 256, 0, stream>>>(
        ffh1b, wtff2, nullptr, 0, ff_b2, 0, nullptr, nullptr, ffhb, 1.f, 512, 2048);
    gemm_t<false,false,false,true,false><<<g512, 256, 0, stream>>>(
        ffhb, wtfsw, nullptr, 0, nullptr, 0, nullptr, shrb, nullptr, 1.f, 512, 512);
    gemm_t<false,false,false,true,false><<<g512, 256, 0, stream>>>(
        ffhb, wtflw, lang, DD, nullptr, 0, nullptr, lngb, nullptr, 1.f, 512, 512);
    gemm_t<true,true,true,true,false><<<g512, 256, 0, stream>>>(
        x2nb, wtfg1, lang, DD, f_g1b, 512, x2n, g1buf, nullptr, 1.f, 512, 512);
    gate_kernel<<<TTOK / 8, 256, 0, stream>>>(g1buf, f_g2w, f_g2b, lang, gateb);
    combine_kernel<<<TD / 1024, 256, 0, stream>>>(shrb, lngb, gateb, x2n, outp);
}
